// Qwen3MoeSparseMoeBlock_33938831573089
// MI455X (gfx1250) — matrix operand order repaired and checked by reading, not compiled, not run on hardware
//
#include <hip/hip_runtime.h>
#include <hip/hip_bf16.h>
#include <stdint.h>

#define T_TOK 512
#define DIM   1024
#define HID   768
#define NEXP  32
#define TOPK  4

typedef __attribute__((ext_vector_type(16))) __bf16 v16bf;
typedef __attribute__((ext_vector_type(8)))  float  v8f;

union FragA { v16bf bf; uint4 q[2]; };
union FragB { v16bf bf; unsigned int u[8]; };

// Pack two fp32 -> two bf16 (truncation) in ONE v_perm_b32:
// result = { hi16(b), hi16(a) }
__device__ __forceinline__ unsigned int pk2bf(float a, float b) {
  return __builtin_amdgcn_perm(__float_as_uint(b), __float_as_uint(a), 0x07060302u);
}
__device__ __forceinline__ unsigned short f2bf(float f) {
  return (unsigned short)(__float_as_uint(f) >> 16);   // truncate
}

// ---------------------------------------------------------------- zero init
__global__ void zero_kernel(float* __restrict__ out, int n, int* __restrict__ counts) {
  int i = blockIdx.x * blockDim.x + threadIdx.x;
  if (i < n) out[i] = 0.0f;
  if (i < NEXP) counts[i] = 0;
}

// ---------------------------------------------------------------- router
// one wave32 per token; lane == expert id
__global__ void router_kernel(const float* __restrict__ h,
                              const float* __restrict__ gate_w,
                              int* __restrict__ counts,
                              int* __restrict__ tokList,
                              float* __restrict__ wList) {
  const int t    = blockIdx.x;
  const int lane = threadIdx.x;             // 0..31 == expert
  const float* hr = h + (size_t)t * DIM;
  const float* gr = gate_w + (size_t)lane * DIM;

  float acc = 0.0f;
  for (int i = 0; i < DIM; i += 4) {
    float4 a = *(const float4*)(hr + i);
    float4 b = *(const float4*)(gr + i);
    acc += a.x * b.x + a.y * b.y + a.z * b.z + a.w * b.w;
  }
  // softmax across the 32 lanes (fp32)
  float m = acc;
  for (int off = 16; off > 0; off >>= 1) m = fmaxf(m, __shfl_xor(m, off));
  float p = __expf(acc - m);
  float s = p;
  for (int off = 16; off > 0; off >>= 1) s += __shfl_xor(s, off);
  float prob = p / s;

  // iterative top-4
  float val = prob;
  float tw[TOPK]; int ti[TOPK]; float tsum = 0.0f;
  for (int k = 0; k < TOPK; ++k) {
    float mv = val;
    for (int off = 16; off > 0; off >>= 1) mv = fmaxf(mv, __shfl_xor(mv, off));
    unsigned long long bm = __ballot(val == mv);
    int src = __ffsll(bm) - 1;
    tw[k] = mv; ti[k] = src; tsum += mv;
    if (lane == src) val = -1.0f;
  }
  if (lane == 0) {
    for (int k = 0; k < TOPK; ++k) {
      int e   = ti[k];
      int pos = atomicAdd(&counts[e], 1);
      tokList[e * T_TOK + pos] = t;
      wList[e * T_TOK + pos]   = tw[k] / tsum;   // renormalized top-k prob
    }
  }
}

// ---------------------------------------------------------------- expert GEMMs
// grid: (expert, token-tile), 256 threads = 8 waves
__global__ void __launch_bounds__(256)
moe_expert_kernel(const float* __restrict__ h,
                  const float* __restrict__ w_gate,
                  const float* __restrict__ w_up,
                  const float* __restrict__ w_down,
                  const int*   __restrict__ counts,
                  const int*   __restrict__ tokList,
                  const float* __restrict__ wList,
                  float* __restrict__ out) {
  __shared__ unsigned short sA[16][DIM];    // 32 KB: token tile, bf16
  __shared__ unsigned short sAct[16][HID];  // 24 KB: SwiGLU activations, bf16
  __shared__ int   sTok[16];
  __shared__ float sW[16];

  const int e    = blockIdx.x;
  const int cnt  = counts[e];
  const int row0 = blockIdx.y * 16;
  if (row0 >= cnt) return;
  const int rows = min(16, cnt - row0);

  const int tid  = threadIdx.x;
  const int lane = tid & 31;
  const int wv   = tid >> 5;

  if (tid < 16) {
    if (tid < rows) {
      sTok[tid] = tokList[e * T_TOK + row0 + tid];
      sW[tid]   = wList[e * T_TOK + row0 + tid];
    } else { sTok[tid] = 0; sW[tid] = 0.0f; }
  }
  __syncthreads();

  // ---- stage 16 x DIM hidden-state tile into LDS as bf16
  for (int i = tid; i < 16 * (DIM / 4); i += 256) {
    int r = i / (DIM / 4);
    int c = (i % (DIM / 4)) * 4;
    float4 v = make_float4(0.f, 0.f, 0.f, 0.f);
    if (r < rows) v = *(const float4*)(h + (size_t)sTok[r] * DIM + c);
    unsigned int* dst = (unsigned int*)&sA[r][c];
    dst[0] = pk2bf(v.x, v.y);
    dst[1] = pk2bf(v.z, v.w);
  }
  __syncthreads();

  // WMMA fragment addressing (ISA 7.12.2, wave32)
  const int mrow   = lane & 15;               // A row (M)
  const int khalfA = (lane < 16) ? 0 : 16;    // A: K sub-block per half-wave
  const int kbaseB = (lane < 16) ? 0 : 16;    // B: K range per half-wave
  const int ncol   = lane & 15;               // B/D column (N)

  const float* Wg = w_gate + (size_t)e * HID * DIM;
  const float* Wu = w_up   + (size_t)e * HID * DIM;

  // ---- gate & up projections: 48 N-tiles over H, 8 waves
  for (int j = wv; j < HID / 16; j += 8) {
    const int n0 = j * 16;
    v8f accG = {}; v8f accU = {};
    const float* wg = Wg + (size_t)(n0 + ncol) * DIM + kbaseB;
    const float* wu = Wu + (size_t)(n0 + ncol) * DIM + kbaseB;
    for (int kc = 0; kc < DIM; kc += 32) {
      FragA a;
      a.q[0] = *(const uint4*)&sA[mrow][kc + khalfA];
      a.q[1] = *(const uint4*)&sA[mrow][kc + 8 + khalfA];
      FragB bg, bu;
      const float* pg = wg + kc;
      const float* pu = wu + kc;
      __builtin_prefetch(pg + 32, 0, 1);   // next K-chunk -> global_prefetch_b8
      __builtin_prefetch(pu + 32, 0, 1);
      #pragma unroll
      for (int q = 0; q < 4; ++q) {
        float4 g4 = *(const float4*)(pg + q * 4);
        float4 u4 = *(const float4*)(pu + q * 4);
        bg.u[q * 2 + 0] = pk2bf(g4.x, g4.y);
        bg.u[q * 2 + 1] = pk2bf(g4.z, g4.w);
        bu.u[q * 2 + 0] = pk2bf(u4.x, u4.y);
        bu.u[q * 2 + 1] = pk2bf(u4.z, u4.w);
      }
      accG = __builtin_amdgcn_wmma_f32_16x16x32_bf16(false, a.bf, false, bg.bf,
                                                     (short)0, accG, false, false);
      accU = __builtin_amdgcn_wmma_f32_16x16x32_bf16(false, a.bf, false, bu.bf,
                                                     (short)0, accU, false, false);
    }
    // SwiGLU: silu(g) * u = g * u * rcp(1 + exp(-g))  (fast v_rcp_f32)
    #pragma unroll
    for (int r = 0; r < 8; ++r) {
      int m = r + ((lane < 16) ? 0 : 8);
      float g = accG[r];
      float u = accU[r];
      float act = g * u * __builtin_amdgcn_rcpf(1.0f + __expf(-g));
      sAct[m][n0 + ncol] = f2bf(act);
    }
  }
  __syncthreads();

  // ---- down projection: 64 N-tiles over D, 8 waves; K = HID
  const float* Wd = w_down + (size_t)e * DIM * HID;
  for (int j = wv; j < DIM / 16; j += 8) {
    const int n0 = j * 16;
    v8f acc = {};
    const float* wd = Wd + (size_t)(n0 + ncol) * HID + kbaseB;
    for (int kc = 0; kc < HID; kc += 32) {
      FragA a;
      a.q[0] = *(const uint4*)&sAct[mrow][kc + khalfA];
      a.q[1] = *(const uint4*)&sAct[mrow][kc + 8 + khalfA];
      FragB b;
      const float* pd = wd + kc;
      __builtin_prefetch(pd + 32, 0, 1);
      #pragma unroll
      for (int q = 0; q < 4; ++q) {
        float4 d4 = *(const float4*)(pd + q * 4);
        b.u[q * 2 + 0] = pk2bf(d4.x, d4.y);
        b.u[q * 2 + 1] = pk2bf(d4.z, d4.w);
      }
      acc = __builtin_amdgcn_wmma_f32_16x16x32_bf16(false, a.bf, false, b.bf,
                                                    (short)0, acc, false, false);
    }
    // weighted combine into output (multiple experts per token -> atomic)
    #pragma unroll
    for (int r = 0; r < 8; ++r) {
      int m = r + ((lane < 16) ? 0 : 8);
      if (m < rows) {
        atomicAdd(&out[(size_t)sTok[m] * DIM + n0 + ncol], acc[r] * sW[m]);
      }
    }
  }
}

// ---------------------------------------------------------------- host entry
extern "C" void kernel_launch(void* const* d_in, const int* in_sizes, int n_in,
                              void* d_out, int out_size, void* d_ws, size_t ws_size,
                              hipStream_t stream) {
  (void)in_sizes; (void)n_in; (void)out_size; (void)ws_size;
  const float* h      = (const float*)d_in[0];  // [1,512,1024]
  const float* gate_w = (const float*)d_in[1];  // [32,1024]
  const float* w_gate = (const float*)d_in[2];  // [32,768,1024]
  const float* w_up   = (const float*)d_in[3];  // [32,768,1024]
  const float* w_down = (const float*)d_in[4];  // [32,1024,768]
  float* out = (float*)d_out;                   // [1,512,1024]

  int*   counts  = (int*)d_ws;
  int*   tokList = counts + NEXP;               // [E][T]
  float* wList   = (float*)(tokList + NEXP * T_TOK); // [E][T]

  const int n = T_TOK * DIM;
  zero_kernel<<<(n + 255) / 256, 256, 0, stream>>>(out, n, counts);
  router_kernel<<<T_TOK, 32, 0, stream>>>(h, gate_w, counts, tokList, wList);
  dim3 grid(NEXP, T_TOK / 16);
  moe_expert_kernel<<<grid, 256, 0, stream>>>(h, w_gate, w_up, w_down,
                                              counts, tokList, wList, out);
}
